// GAT_25383256720122
// MI455X (gfx1250) — matrix operand order repaired and checked by reading, not compiled, not run on hardware
//
#include <hip/hip_runtime.h>
#include <cstdint>
#include <cstddef>

// ---------------------------------------------------------------------------
// GAT forward for MI455X (gfx1250, wave32). GEMMs use v_wmma_f32_16x16x32_bf16
// with weights staged bf16-swizzled in LDS; edge softmax/scatter uses f32
// global atomics (the bandwidth-bound part of this graph workload).
// ---------------------------------------------------------------------------

#define NEG_SLOPE 0.2f

typedef __attribute__((ext_vector_type(16))) __bf16 v16bf;
typedef __attribute__((ext_vector_type(8)))  float  v8f;

// order-preserving float <-> uint encoding for atomicMax-based segment max
__device__ __forceinline__ unsigned enc_ord(float f) {
  unsigned u = __float_as_uint(f);
  return (u & 0x80000000u) ? ~u : (u | 0x80000000u);
}
__device__ __forceinline__ float dec_ord(unsigned k) {
  unsigned u = (k & 0x80000000u) ? (k & 0x7FFFFFFFu) : ~k;
  return __uint_as_float(u);
}

// ---------------------------------------------------------------------------
// OUT[nrows,128] = act(A[nrows,KIN] @ W[KIN,128] (+ bias))
// Block = 256 threads = 8 waves; each wave owns a 16-row slab, all 128 cols.
// W is staged in LDS as bf16, pre-swizzled into WMMA B-fragment layout:
//   frag(kc,nt): lane l, elem i -> W[kc*32 + ((l&16)>>1) + (i&7) + (i&8)*2][nt*16 + (l&15)]
// A fragment per-lane layout (16-bit A 16x32):
//   lane l, elem i -> A[row0+(l&15)][kc*32 + ((l&16)>>1) + (i&7) + (i&8)*2]
// ---------------------------------------------------------------------------
template <int KIN>
__global__ __launch_bounds__(256) void k_gemm_wmma(
    const float* __restrict__ A, const float* __restrict__ W,
    const float* __restrict__ bias, float* __restrict__ OUT,
    int nrows, int relu)
{
  extern __shared__ __bf16 wlds[];  // KIN*128 bf16
  const int tid  = threadIdx.x;
  const int lane = tid & 31;
  const int wave = tid >> 5;
  constexpr int KC = KIN / 32;

  // stage + swizzle weights into LDS (one pass, coalesced-ish reads)
  for (int idx = tid; idx < KIN * 128; idx += 256) {
    const int f  = idx >> 9;       // frag id = kc*8 + nt (512 bf16 per frag)
    const int r  = idx & 511;
    const int l  = r >> 4;
    const int i  = r & 15;
    const int kc = f >> 3;
    const int nt = f & 7;
    const int k  = kc * 32 + ((l & 16) >> 1) + (i & 7) + (i & 8) * 2;
    const int n  = nt * 16 + (l & 15);
    wlds[idx] = (__bf16)W[(size_t)k * 128 + n];
  }
  __syncthreads();

  const int row0 = blockIdx.x * 128 + wave * 16;
  const int m    = row0 + (lane & 15);

  v8f acc[8];
#pragma unroll
  for (int nt = 0; nt < 8; ++nt)
#pragma unroll
    for (int r = 0; r < 8; ++r) acc[nt][r] = 0.0f;

#pragma unroll
  for (int kc = 0; kc < KC; ++kc) {
    v16bf a;
    const int kbase = kc * 32 + ((lane & 16) >> 1);
    if (m < nrows) {
      const float* ap = A + (size_t)m * KIN + kbase;
#pragma unroll
      for (int i = 0; i < 8; ++i) a[i]     = (__bf16)ap[i];
#pragma unroll
      for (int i = 0; i < 8; ++i) a[8 + i] = (__bf16)ap[16 + i];
    } else {
#pragma unroll
      for (int i = 0; i < 16; ++i) a[i] = (__bf16)0.0f;
    }
#pragma unroll
    for (int nt = 0; nt < 8; ++nt) {
      const v16bf b = *(const v16bf*)(&wlds[((kc * 8 + nt) << 9) + (lane << 4)]);
      acc[nt] = __builtin_amdgcn_wmma_f32_16x16x32_bf16(
          false, a, false, b, (short)0, acc[nt], false, false);
    }
  }

  // epilogue: C/D layout -> lane l, VGPR r: row = r + (l>=16?8:0), col = l&15
  const int rowh = (lane & 16) >> 1;
#pragma unroll
  for (int nt = 0; nt < 8; ++nt) {
    const int col = nt * 16 + (lane & 15);
    const float bv = bias ? bias[col] : 0.0f;
#pragma unroll
    for (int r = 0; r < 8; ++r) {
      const int row = row0 + rowh + r;
      if (row < nrows) {
        float v = acc[nt][r] + bv;
        if (relu) v = v > 0.0f ? v : 0.0f;
        OUT[(size_t)row * 128 + col] = v;
      }
    }
  }
}

// ---------------------------------------------------------------------------
// small helpers
// ---------------------------------------------------------------------------
__global__ void k_fill_f32(float* __restrict__ p, float v, long long n) {
  long long t = (long long)blockIdx.x * blockDim.x + threadIdx.x;
  if (t < n) p[t] = v;
}
__global__ void k_fill_u32(unsigned* __restrict__ p, unsigned v, long long n) {
  long long t = (long long)blockIdx.x * blockDim.x + threadIdx.x;
  if (t < n) p[t] = v;
}

// reprs[batch[n]] += h[n]  (h already non-negative here)
__global__ void k_segadd(const float* __restrict__ h, const int* __restrict__ batch,
                         float* __restrict__ reprs, long long n) {
  long long t = (long long)blockIdx.x * blockDim.x + threadIdx.x;
  if (t >= n) return;
  const int node = (int)(t >> 7), f = (int)(t & 127);
  atomicAdd(&reprs[(size_t)batch[node] * 128 + f], h[t]);
}

// h = relu(h + bias) in place, then reprs[batch[n]] += h[n]
__global__ void k_bias_relu_segadd(float* __restrict__ h, const float* __restrict__ bias,
                                   const int* __restrict__ batch,
                                   float* __restrict__ reprs, long long n) {
  long long t = (long long)blockIdx.x * blockDim.x + threadIdx.x;
  if (t >= n) return;
  const int node = (int)(t >> 7), f = (int)(t & 127);
  float v = h[t] + bias[f];
  v = v > 0.0f ? v : 0.0f;
  h[t] = v;
  atomicAdd(&reprs[(size_t)batch[node] * 128 + f], v);
}

// alpha_src[n,head] = <h[n,head,:], a_src[head,:]> ; same for dst
__global__ void k_alpha(const float* __restrict__ h, const float* __restrict__ asrc,
                        const float* __restrict__ adst,
                        float* __restrict__ aS, float* __restrict__ aD, int N) {
  int t = blockIdx.x * blockDim.x + threadIdx.x;
  if (t >= N * 2) return;
  const int n = t >> 1, head = t & 1;
  const float* hp = h + (size_t)n * 128 + head * 64;
  const float* sp = asrc + head * 64;
  const float* dp = adst + head * 64;
  float s = 0.0f, d = 0.0f;
#pragma unroll 4
  for (int c = 0; c < 64; ++c) {
    const float hv = hp[c];
    s += hv * sp[c];
    d += hv * dp[c];
  }
  aS[t] = s;
  aD[t] = d;
}

__device__ __forceinline__ void edge_sd(const int* __restrict__ ei, int E, int e,
                                        int& s, int& d) {
  if (e < E) { s = ei[e]; d = ei[E + e]; }   // self-loops appended
  else       { s = d = e - E; }
}

__global__ void k_edge_max(const int* __restrict__ ei, int E, int ET,
                           const float* __restrict__ aS, const float* __restrict__ aD,
                           unsigned* __restrict__ mkey) {
  int t = blockIdx.x * blockDim.x + threadIdx.x;
  if (t >= ET * 2) return;
  const int e = t >> 1, head = t & 1;
  int s, d; edge_sd(ei, E, e, s, d);
  float v = aS[s * 2 + head] + aD[d * 2 + head];
  v = v > 0.0f ? v : NEG_SLOPE * v;
  atomicMax(&mkey[d * 2 + head], enc_ord(v));
}

__global__ void k_edge_exp(const int* __restrict__ ei, int E, int ET,
                           const float* __restrict__ aS, const float* __restrict__ aD,
                           const unsigned* __restrict__ mkey,
                           float* __restrict__ exbuf, float* __restrict__ denom) {
  int t = blockIdx.x * blockDim.x + threadIdx.x;
  if (t >= ET * 2) return;
  const int e = t >> 1, head = t & 1;
  int s, d; edge_sd(ei, E, e, s, d);
  float v = aS[s * 2 + head] + aD[d * 2 + head];
  v = v > 0.0f ? v : NEG_SLOPE * v;
  const float ex = __expf(v - dec_ord(mkey[d * 2 + head]));
  exbuf[t] = ex;
  atomicAdd(&denom[d * 2 + head], ex);
}

// one wave per edge: out[dst] += alpha * h[src]; lane f strides of 32
__global__ __launch_bounds__(256) void k_edge_scatter(
    const int* __restrict__ ei, int E, int ET,
    const float* __restrict__ ht, const float* __restrict__ exbuf,
    const float* __restrict__ denom, float* __restrict__ outp) {
  const int e = blockIdx.x * 8 + (threadIdx.x >> 5);
  if (e >= ET) return;
  const int lane = threadIdx.x & 31;
  int s, d; edge_sd(ei, E, e, s, d);
  const float w0 = exbuf[e * 2 + 0] / denom[d * 2 + 0];
  const float w1 = exbuf[e * 2 + 1] / denom[d * 2 + 1];
  const float* hp = ht + (size_t)s * 128;
  float* op = outp + (size_t)d * 128;
#pragma unroll
  for (int j = 0; j < 4; ++j) {
    const int f = lane + 32 * j;
    const float wgt = (f < 64) ? w0 : w1;
    atomicAdd(&op[f], hp[f] * wgt);
  }
}

// final tiny layer + log_softmax: one thread per graph
__global__ void k_post2(const float* __restrict__ z1, const float* __restrict__ w,
                        const float* __restrict__ b, float* __restrict__ out, int G) {
  const int g = blockIdx.x * blockDim.x + threadIdx.x;
  if (g >= G) return;
  float z[10];
#pragma unroll
  for (int c = 0; c < 10; ++c) z[c] = b[c];
  const float* zp = z1 + (size_t)g * 128;
  for (int k = 0; k < 128; ++k) {
    const float zv = zp[k];
#pragma unroll
    for (int c = 0; c < 10; ++c) z[c] += zv * w[k * 10 + c];
  }
  float m = z[0];
#pragma unroll
  for (int c = 1; c < 10; ++c) m = fmaxf(m, z[c]);
  float se = 0.0f;
#pragma unroll
  for (int c = 0; c < 10; ++c) se += __expf(z[c] - m);
  const float lse = m + __logf(se);
#pragma unroll
  for (int c = 0; c < 10; ++c) out[g * 10 + c] = z[c] - lse;
}

// ---------------------------------------------------------------------------
extern "C" void kernel_launch(void* const* d_in, const int* in_sizes, int n_in,
                              void* d_out, int out_size, void* d_ws, size_t ws_size,
                              hipStream_t stream) {
  const float* x     = (const float*)d_in[0];
  const int*   ei    = (const int*)d_in[1];
  const int*   batch = (const int*)d_in[2];
  const float* pre_w = (const float*)d_in[3];
  const float* pre_b = (const float*)d_in[4];
  const float* w1    = (const float*)d_in[5];
  const float* asrc1 = (const float*)d_in[6];
  const float* adst1 = (const float*)d_in[7];
  const float* b1    = (const float*)d_in[8];
  const float* w2    = (const float*)d_in[9];
  const float* asrc2 = (const float*)d_in[10];
  const float* adst2 = (const float*)d_in[11];
  const float* b2    = (const float*)d_in[12];
  const float* pw1   = (const float*)d_in[13];
  const float* pb1   = (const float*)d_in[14];
  const float* pw2   = (const float*)d_in[15];
  const float* pb2   = (const float*)d_in[16];
  (void)n_in; (void)ws_size;

  const int N  = in_sizes[0] / 256;
  const int E  = in_sizes[1] / 2;
  const int ET = E + N;
  const int G  = out_size / 10;

  // carve workspace
  char* ws = (char*)d_ws;
  size_t off = 0;
  auto carve = [&](size_t bytes) -> char* {
    char* p = ws + off;
    off = (off + bytes + 255) & ~(size_t)255;
    return p;
  };
  float*    hA    = (float*)carve((size_t)N * 128 * 4);
  float*    hB    = (float*)carve((size_t)N * 128 * 4);
  float*    aS    = (float*)carve((size_t)N * 2 * 4);
  float*    aD    = (float*)carve((size_t)N * 2 * 4);
  unsigned* mkey  = (unsigned*)carve((size_t)N * 2 * 4);
  float*    denom = (float*)carve((size_t)N * 2 * 4);
  float*    exbuf = (float*)carve((size_t)ET * 2 * 4);
  float*    reprs = (float*)carve((size_t)G * 128 * 4);
  float*    z1    = (float*)carve((size_t)G * 128 * 4);

  const long long NH = (long long)N * 128;
  dim3 b256(256);
  auto gridN = [](long long n) { return dim3((unsigned)((n + 255) / 256)); };

  // pre-layer: hA = relu(x @ pre_w + pre_b)
  k_gemm_wmma<256><<<dim3((N + 127) / 128), b256, 256 * 128 * 2, stream>>>(
      x, pre_w, pre_b, hA, N, 1);
  k_fill_f32<<<gridN((long long)G * 128), b256, 0, stream>>>(reprs, 0.0f, (long long)G * 128);
  k_segadd<<<gridN(NH), b256, 0, stream>>>(hA, batch, reprs, NH);

  for (int layer = 0; layer < 2; ++layer) {
    const float* w  = layer ? w2 : w1;
    const float* as = layer ? asrc2 : asrc1;
    const float* ad = layer ? adst2 : adst1;
    const float* bb = layer ? b2 : b1;

    // hB = hA @ w  (hA dead afterwards -> reuse as aggregation target)
    k_gemm_wmma<128><<<dim3((N + 127) / 128), b256, 128 * 128 * 2, stream>>>(
        hA, w, nullptr, hB, N, 0);
    k_alpha<<<gridN((long long)N * 2), b256, 0, stream>>>(hB, as, ad, aS, aD, N);

    k_fill_u32<<<gridN((long long)N * 2), b256, 0, stream>>>(mkey, 0u, (long long)N * 2);
    k_fill_f32<<<gridN((long long)N * 2), b256, 0, stream>>>(denom, 0.0f, (long long)N * 2);
    k_fill_f32<<<gridN(NH), b256, 0, stream>>>(hA, 0.0f, NH);

    k_edge_max<<<gridN((long long)ET * 2), b256, 0, stream>>>(ei, E, ET, aS, aD, mkey);
    k_edge_exp<<<gridN((long long)ET * 2), b256, 0, stream>>>(ei, E, ET, aS, aD, mkey,
                                                              exbuf, denom);
    k_edge_scatter<<<dim3((ET + 7) / 8), b256, 0, stream>>>(ei, E, ET, hB, exbuf,
                                                            denom, hA);
    k_bias_relu_segadd<<<gridN(NH), b256, 0, stream>>>(hA, bb, batch, reprs, NH);
  }

  // head: z1 = relu(reprs @ pw1 + pb1); out = log_softmax(z1 @ pw2 + pb2)
  k_gemm_wmma<128><<<dim3((G + 127) / 128), b256, 128 * 128 * 2, stream>>>(
      reprs, pw1, pb1, z1, G, 1);
  k_post2<<<dim3((G + 127) / 128), dim3(128), 0, stream>>>(z1, pw2, pb2,
                                                           (float*)d_out, G);
}
